// Cabasc_50560355008544
// MI455X (gfx1250) — matrix operand order repaired and checked by reading, not compiled, not run on hardware
//
#include <hip/hip_runtime.h>
#include <hip/hip_bf16.h>

typedef __attribute__((ext_vector_type(16))) __bf16 v16bf;
typedef __attribute__((ext_vector_type(8)))  __bf16 v8bf;
typedef __attribute__((ext_vector_type(8)))  float  v8f;

#define B_   64
#define S_   512
#define TA_  8
#define D_   768
#define DPAD 776   // 768 + 8 halves pad -> LDS rows start 4 banks apart per lane

// Branch-free tanh: clamp, then 1 - 2/(exp(2x)+1) with native exp/rcp.
// Avoids libm tanhf's divergent exec-mask branches in the hot loop.
__device__ __forceinline__ float fast_tanh(float x) {
    float xc = fminf(fmaxf(x, -15.f), 15.f);
    float e  = __expf(2.f * xc);
    float r  = __builtin_amdgcn_rcpf(e + 1.f);
    return fmaf(-2.f, r, 1.f);
}

// ---------------- Kernel 1: per-batch lengths + v_a, v_s ----------------
__global__ void reduce_kernel(const float* __restrict__ memory,
                              const float* __restrict__ aspect,
                              const int* __restrict__ ids,
                              const int* __restrict__ tids,
                              float* __restrict__ mlen, float* __restrict__ alen,
                              float* __restrict__ v_s, float* __restrict__ v_a) {
    int b = blockIdx.x;
    int t = threadIdx.x;
    __shared__ float r1[256], r2[256];
    float c1 = 0.f, c2 = 0.f;
    for (int s = t; s < S_;  s += 256) c1 += (ids [b*S_ + s] != 0) ? 1.f : 0.f;
    for (int s = t; s < TA_; s += 256) c2 += (tids[b*TA_ + s] != 0) ? 1.f : 0.f;
    r1[t] = c1; r2[t] = c2;
    __syncthreads();
    for (int off = 128; off > 0; off >>= 1) {
        if (t < off) { r1[t] += r1[t+off]; r2[t] += r2[t+off]; }
        __syncthreads();
    }
    float ml = r1[0], al = r2[0];
    if (t == 0) { mlen[b] = ml; alen[b] = al; }
    for (int d = t; d < D_; d += 256) {
        float acc = 0.f;
        for (int s = 0; s < S_; ++s) acc += memory[((size_t)b*S_ + s)*D_ + d];
        v_s[b*D_ + d] = acc / ml;
        float acca = 0.f;
        for (int s = 0; s < TA_; ++s) acca += aspect[((size_t)b*TA_ + s)*D_ + d];
        v_a[b*D_ + d] = acca / al;
    }
}

// ---------------- Kernel 2: ctx = v_a @ W1_a + v_s @ W1_s + b1 ----------------
__global__ void ctx_kernel(const float* __restrict__ v_a,
                           const float* __restrict__ v_s,
                           const float* __restrict__ W1,
                           const float* __restrict__ b1,
                           float* __restrict__ ctx) {
    int b = blockIdx.x;
    int e = blockIdx.y * 256 + threadIdx.x;
    const float* Wa = W1 + (size_t)D_ * D_;
    const float* Ws = W1 + (size_t)2 * D_ * D_;
    float acc = b1[e];
    for (int d = 0; d < D_; ++d) {
        acc += v_a[b*D_ + d] * Wa[(size_t)d*D_ + e]
             + v_s[b*D_ + d] * Ws[(size_t)d*D_ + e];
    }
    ctx[b*D_ + e] = acc;
}

// ---------------- Kernel 3: Wt[e][d] = bf16(W1_m[d][e])  (B-operand layout) ----------------
__global__ void wt_kernel(const float* __restrict__ W1, __bf16* __restrict__ Wt) {
    int idx = blockIdx.x * 256 + threadIdx.x;  // idx = e*768 + d
    int e = idx / D_;
    int d = idx - e * D_;
    Wt[idx] = (__bf16)W1[(size_t)d * D_ + e];
}

// ---------------- Kernel 4: scores[b,s] = tanh(mem_loc @ W1_m + ctx) . w2  (WMMA bf16) ----------------
__global__ void __launch_bounds__(128) scores_kernel(
        const float*  __restrict__ memory,
        const __bf16* __restrict__ Wt,
        const float*  __restrict__ ctx,
        const float*  __restrict__ w2,
        const float*  __restrict__ mlen,
        float* __restrict__ scores) {
    __shared__ __align__(16) __bf16 Alds[64][DPAD];

    const int t  = threadIdx.x;
    const int m0 = blockIdx.x * 64;       // 64 flattened (b,s) rows per block
    const int b  = m0 / S_;
    const float ml = mlen[b];

    // Stage 64 rows of loc-scaled memory as bf16 into LDS (fully coalesced global reads)
    for (int i = t; i < 64 * D_; i += 128) {
        int row = i / D_;
        int s   = (m0 - b * S_) + row;
        float loc = ((float)s < ml) ? (1.f - (float)s / ml) : 1.f;
        Alds[row][i - row * D_] = (__bf16)(memory[(size_t)m0 * D_ + i] * loc);
    }
    __syncthreads();

    const int lane = t & 31;
    const int wave = t >> 5;      // 4 waves, 16 rows each
    const int half = lane >> 4;   // lane group 0-15 / 16-31
    const int l15  = lane & 15;

    float sc[8];
#pragma unroll
    for (int j = 0; j < 8; ++j) sc[j] = 0.f;

    const __bf16* arow = &Alds[wave * 16 + l15][half * 8];

    for (int nt = 0; nt < D_ / 16; ++nt) {
        const int e0 = nt * 16;
        const float cinit = ctx[b * D_ + e0 + l15];   // C layout: N = lane%16, same for all 8 VGPRs
        v8f acc;
#pragma unroll
        for (int j = 0; j < 8; ++j) acc[j] = cinit;

        const __bf16* brow = Wt + (size_t)(e0 + l15) * D_ + half * 8;

#pragma unroll 4
        for (int k = 0; k < D_; k += 32) {
            v16bf a, bm;
            // A 16x32 bf16 layout: lane<16 -> K = k+{0..7, 16..23}; lane>=16 -> +8
            ((v8bf*)&a)[0]  = *(const v8bf*)(arow + k);
            ((v8bf*)&a)[1]  = *(const v8bf*)(arow + k + 16);
            // B 32x16 bf16 layout: lane n = column n, K = k + half*8 + {0..7, 16..23}
            ((v8bf*)&bm)[0] = *(const v8bf*)(brow + k);
            ((v8bf*)&bm)[1] = *(const v8bf*)(brow + k + 16);
            acc = __builtin_amdgcn_wmma_f32_16x16x32_bf16(
                      false, a, false, bm, (short)0, acc, false, false);
        }
        const float w2e = w2[e0 + l15];
#pragma unroll
        for (int j = 0; j < 8; ++j) sc[j] += fast_tanh(acc[j]) * w2e;
    }

    // Sum over the 16 lanes holding the same row (xor 1,2,4,8 stays within half-wave)
#pragma unroll
    for (int m = 1; m < 16; m <<= 1) {
#pragma unroll
        for (int j = 0; j < 8; ++j) sc[j] += __shfl_xor(sc[j], m, 32);
    }
    if (l15 == 0) {
#pragma unroll
        for (int j = 0; j < 8; ++j)
            scores[m0 + wave * 16 + half * 8 + j] = sc[j];
    }
}

// ---------------- Kernel 5: softmax + v_ts + v_s -> v_ns ----------------
__global__ void softmax_vts_kernel(const float* __restrict__ memory,
                                   const float* __restrict__ scores,
                                   const float* __restrict__ v_s,
                                   const float* __restrict__ mlen,
                                   float* __restrict__ v_ns) {
    int b = blockIdx.x, t = threadIdx.x;
    __shared__ float sc[S_];
    __shared__ float red[256];
    __shared__ float s_inv;
    float ml = mlen[b];
    float a0 = scores[b*S_ + t], a1 = scores[b*S_ + t + 256];
    red[t] = fmaxf(a0, a1);
    __syncthreads();
    for (int off = 128; off > 0; off >>= 1) {
        if (t < off) red[t] = fmaxf(red[t], red[t+off]);
        __syncthreads();
    }
    float mx = red[0];
    __syncthreads();
    float e0 = expf(a0 - mx), e1 = expf(a1 - mx);
    sc[t] = e0; sc[t + 256] = e1;
    red[t] = e0 + e1;
    __syncthreads();
    for (int off = 128; off > 0; off >>= 1) {
        if (t < off) red[t] += red[t+off];
        __syncthreads();
    }
    if (t == 0) s_inv = 1.f / red[0];
    __syncthreads();
    float inv = s_inv;
    for (int d = t; d < D_; d += 256) {
        float acc = 0.f;
        for (int s = 0; s < S_; ++s) {
            float loc = ((float)s < ml) ? (1.f - (float)s / ml) : 1.f;
            acc += sc[s] * inv * memory[((size_t)b*S_ + s)*D_ + d] * loc;
        }
        v_ns[b*D_ + d] = acc + v_s[b*D_ + d];
    }
}

// ---------------- Kernel 6: v_ms = tanh(v_ns @ Wm + bm) ----------------
__global__ void vms_kernel(const float* __restrict__ v_ns,
                           const float* __restrict__ Wm,
                           const float* __restrict__ bm,
                           float* __restrict__ v_ms) {
    int b = blockIdx.x;
    int e = blockIdx.y * 256 + threadIdx.x;
    float acc = bm[e];
    for (int d = 0; d < D_; ++d) acc += v_ns[b*D_ + d] * Wm[(size_t)d*D_ + e];
    v_ms[b*D_ + e] = fast_tanh(acc);
}

// ---------------- Kernel 7: logits = v_ms @ Wd + bd ----------------
__global__ void logits_kernel(const float* __restrict__ v_ms,
                              const float* __restrict__ Wd,
                              const float* __restrict__ bd,
                              float* __restrict__ out) {
    int t = blockIdx.x * blockDim.x + threadIdx.x;
    if (t >= B_ * 3) return;
    int b = t / 3, j = t - b * 3;
    float acc = bd[j];
    for (int d = 0; d < D_; ++d) acc += v_ms[b*D_ + d] * Wd[d*3 + j];
    out[t] = acc;
}

extern "C" void kernel_launch(void* const* d_in, const int* in_sizes, int n_in,
                              void* d_out, int out_size, void* d_ws, size_t ws_size,
                              hipStream_t stream) {
    const float* memory = (const float*)d_in[0];
    const float* aspect = (const float*)d_in[1];
    const int*   ids    = (const int*)d_in[2];
    const int*   tids   = (const int*)d_in[3];
    const float* W1     = (const float*)d_in[4];
    const float* b1     = (const float*)d_in[5];
    const float* w2     = (const float*)d_in[6];
    const float* Wm     = (const float*)d_in[7];
    const float* bm     = (const float*)d_in[8];
    const float* Wd     = (const float*)d_in[9];
    const float* bd     = (const float*)d_in[10];
    float* out = (float*)d_out;

    float* ws = (float*)d_ws;
    float* mlen   = ws;  ws += B_;
    float* alen   = ws;  ws += B_;
    float* v_a    = ws;  ws += B_ * D_;
    float* v_s    = ws;  ws += B_ * D_;
    float* ctx    = ws;  ws += B_ * D_;
    float* scores = ws;  ws += B_ * S_;
    float* v_ns   = ws;  ws += B_ * D_;
    float* v_ms   = ws;  ws += B_ * D_;
    __bf16* Wt    = (__bf16*)ws;           // 768*768 bf16, 16B-aligned offset

    reduce_kernel<<<B_, 256, 0, stream>>>(memory, aspect, ids, tids, mlen, alen, v_s, v_a);
    ctx_kernel<<<dim3(B_, 3), 256, 0, stream>>>(v_a, v_s, W1, b1, ctx);
    wt_kernel<<<(D_ * D_) / 256, 256, 0, stream>>>(W1, Wt);
    scores_kernel<<<(B_ * S_) / 64, 128, 0, stream>>>(memory, Wt, ctx, w2, mlen, scores);
    softmax_vts_kernel<<<B_, 256, 0, stream>>>(memory, scores, v_s, mlen, v_ns);
    vms_kernel<<<dim3(B_, 3), 256, 0, stream>>>(v_ns, Wm, bm, v_ms);
    logits_kernel<<<1, 256, 0, stream>>>(v_ms, Wd, bd, out);
}
